// MultiSelfAttention_79070347920114
// MI455X (gfx1250) — compile-verified, matrix operand order checked by reading, not run on hardware
//
#include <hip/hip_runtime.h>
#include <hip/hip_bf16.h>

#define HEADS 16
#define DMODEL 1024
#define DK 64
#define SEQ 2048
#define BATCH 4

typedef __attribute__((ext_vector_type(16))) __bf16 v16bf;
typedef __attribute__((ext_vector_type(8)))  float  v8f;
typedef __attribute__((ext_vector_type(8)))  unsigned short ushort8v;
typedef __attribute__((ext_vector_type(4)))  unsigned short ushort4v;

union FragBF { v16bf v; ushort8v h[2]; unsigned short u[16]; };
union AccF   { v8f v; float f[8]; };

__device__ __forceinline__ unsigned short f2bf(float x) {
  union { float f; unsigned int u; } c; c.f = x;
  unsigned int r = c.u + 0x7FFFu + ((c.u >> 16) & 1u);   // round-to-nearest-even
  return (unsigned short)(r >> 16);
}

__device__ __forceinline__ v8f wmma_bf16(v16bf a, v16bf b, v8f c) {
  return __builtin_amdgcn_wmma_f32_16x16x32_bf16(false, a, false, b, (short)0, c,
                                                 false, false);
}

// ---------------------------------------------------------------------------
// Tiled GEMM: Y[m,n] = sum_k X[m,k] * W[n,k] + bias[n]
//   MODE 0: Y -> bf16, layout [B, H, S, DK]   (head-split Q/K)
//   MODE 2: Y -> bf16, layout [B, H, DK, S]   (head-split transposed V)
//   MODE 1: Y -> f32,  layout [M, DMODEL]     (final output projection)
//   IN_BF16: X is bf16 (ushort) instead of f32
// Block: 256 threads (8 waves), tile 128x128, Kc=64 staged in LDS as bf16.
// ---------------------------------------------------------------------------
template<int MODE, bool IN_BF16>
__global__ __launch_bounds__(256)
void gemm_wmma_kernel(const void* __restrict__ Xv, const float* __restrict__ W,
                      const float* __restrict__ bias, void* __restrict__ Out)
{
  constexpr int LDP = 80;                 // padded LDS row stride (ushorts)
  __shared__ unsigned short As[128 * LDP];
  __shared__ unsigned short Bs[128 * LDP];

  const int tid  = threadIdx.x;
  const int lane = tid & 31;
  const int wave = tid >> 5;
  const int wm   = wave & 3;              // M group of 32 rows
  const int wn   = wave >> 2;             // N group of 64 cols
  const int blockM = blockIdx.x * 128;
  const int blockN = blockIdx.y * 128;
  const int r    = lane & 15;
  const int half = lane >> 4;

  AccF acc[2][4];
#pragma unroll
  for (int mi = 0; mi < 2; ++mi)
#pragma unroll
    for (int ni = 0; ni < 4; ++ni)
#pragma unroll
      for (int e = 0; e < 8; ++e) acc[mi][ni].f[e] = 0.f;

  const int lrow = tid >> 1;              // 0..127
  const int lcol = (tid & 1) * 32;        // 0 or 32

  for (int kt = 0; kt < DMODEL; kt += 64) {
    // ---- stage X tile (convert f32 -> bf16 in registers) ----
    if (IN_BF16) {
      const unsigned short* src =
          (const unsigned short*)Xv + (size_t)(blockM + lrow) * DMODEL + kt + lcol;
#pragma unroll
      for (int v2 = 0; v2 < 4; ++v2)
        *(ushort8v*)&As[lrow * LDP + lcol + v2 * 8] = *(const ushort8v*)(src + v2 * 8);
    } else {
      const float* src = (const float*)Xv + (size_t)(blockM + lrow) * DMODEL + kt + lcol;
#pragma unroll
      for (int v2 = 0; v2 < 8; ++v2) {
        float4 t = *(const float4*)(src + v2 * 4);
        ushort4v p;
        p[0] = f2bf(t.x); p[1] = f2bf(t.y); p[2] = f2bf(t.z); p[3] = f2bf(t.w);
        *(ushort4v*)&As[lrow * LDP + lcol + v2 * 4] = p;
      }
    }
    // ---- stage W tile (rows are output features n; k-contiguous) ----
    {
      const float* src = W + (size_t)(blockN + lrow) * DMODEL + kt + lcol;
#pragma unroll
      for (int v2 = 0; v2 < 8; ++v2) {
        float4 t = *(const float4*)(src + v2 * 4);
        ushort4v p;
        p[0] = f2bf(t.x); p[1] = f2bf(t.y); p[2] = f2bf(t.z); p[3] = f2bf(t.w);
        *(ushort4v*)&Bs[lrow * LDP + lcol + v2 * 4] = p;
      }
    }
    __syncthreads();

#pragma unroll
    for (int k0 = 0; k0 < 64; k0 += 32) {
      const int kb = k0 + (half << 3);    // A-layout: half-wave K offset
      FragBF afr[2];
#pragma unroll
      for (int mi = 0; mi < 2; ++mi) {
        const int row = wm * 32 + mi * 16 + r;
        afr[mi].h[0] = *(const ushort8v*)&As[row * LDP + kb];
        afr[mi].h[1] = *(const ushort8v*)&As[row * LDP + kb + 16];
      }
      FragBF bfr[4];
#pragma unroll
      for (int ni = 0; ni < 4; ++ni) {
        const int row = wn * 64 + ni * 16 + r;
        bfr[ni].h[0] = *(const ushort8v*)&Bs[row * LDP + kb];
        bfr[ni].h[1] = *(const ushort8v*)&Bs[row * LDP + kb + 16];
      }
#pragma unroll
      for (int mi = 0; mi < 2; ++mi)
#pragma unroll
        for (int ni = 0; ni < 4; ++ni)
          acc[mi][ni].v = wmma_bf16(afr[mi].v, bfr[ni].v, acc[mi][ni].v);
    }
    __syncthreads();
  }

  // ---- epilogue: bias add + store (C layout: row = e + 8*half, col = r) ----
#pragma unroll
  for (int mi = 0; mi < 2; ++mi) {
#pragma unroll
    for (int ni = 0; ni < 4; ++ni) {
      const int gn = blockN + wn * 64 + ni * 16 + r;
      const float bv = bias[gn];
#pragma unroll
      for (int e = 0; e < 8; ++e) {
        const int gm = blockM + wm * 32 + mi * 16 + e + (half << 3);
        const float val = acc[mi][ni].f[e] + bv;
        if (MODE == 1) {
          ((float*)Out)[(size_t)gm * DMODEL + gn] = val;
        } else {
          const int b = gm >> 11, s = gm & (SEQ - 1);
          const int h = gn >> 6,  d = gn & (DK - 1);
          if (MODE == 0)
            ((unsigned short*)Out)[(((size_t)b * HEADS + h) * SEQ + s) * DK + d] = f2bf(val);
          else  // MODE 2: V transposed per head
            ((unsigned short*)Out)[(((size_t)b * HEADS + h) * DK + d) * SEQ + s] = f2bf(val);
        }
      }
    }
  }
}

// ---------------------------------------------------------------------------
// Flash attention per (query-tile, batch*head). Block: 256 threads (8 waves),
// each wave owns 16 query rows. Streams 128-key blocks; scores and PV both
// on bf16 WMMA; online softmax in f32 with 16-lane shfl reductions.
// ---------------------------------------------------------------------------
__global__ __launch_bounds__(256)
void flash_attn_kernel(const unsigned short* __restrict__ Qh,
                       const unsigned short* __restrict__ Kh,
                       const unsigned short* __restrict__ Vt,
                       unsigned short* __restrict__ ctx)
{
  __shared__ unsigned short Plds[8 * 16 * 128];   // per-wave P slab (32 KB)

  const int bh = blockIdx.y;
  const int b  = bh >> 4;
  const int h  = bh & 15;
  const int qbase = blockIdx.x * 128;
  const int tid  = threadIdx.x;
  const int lane = tid & 31;
  const int wave = tid >> 5;
  const int r    = lane & 15;
  const int half = lane >> 4;

  // Q A-fragments for this wave's 16 rows (Dk=64 -> 2 K-steps), direct from global
  FragBF qf[2];
  const unsigned short* Qp = Qh + ((size_t)bh * SEQ + qbase + wave * 16) * DK;
#pragma unroll
  for (int ks = 0; ks < 2; ++ks) {
    const int kb = ks * 32 + (half << 3);
    qf[ks].h[0] = *(const ushort8v*)&Qp[r * DK + kb];
    qf[ks].h[1] = *(const ushort8v*)&Qp[r * DK + kb + 16];
  }

  AccF o[4];
  float rmax[8], rsum[8];
#pragma unroll
  for (int dt = 0; dt < 4; ++dt)
#pragma unroll
    for (int e = 0; e < 8; ++e) o[dt].f[e] = 0.f;
#pragma unroll
  for (int e = 0; e < 8; ++e) { rmax[e] = -1e30f; rsum[e] = 0.f; }

  unsigned short* Pw = &Plds[wave * 16 * 128];
  const float scale = 0.125f;   // 1/sqrt(64)

  for (int kt = 0; kt < SEQ; kt += 128) {
    const unsigned short* Kp = Kh + ((size_t)bh * SEQ + kt) * DK;

    // ---- scores S = Q K^T (16 x 128) ----
    AccF sc[8];
#pragma unroll
    for (int nt = 0; nt < 8; ++nt)
#pragma unroll
      for (int e = 0; e < 8; ++e) sc[nt].f[e] = 0.f;

#pragma unroll
    for (int ks = 0; ks < 2; ++ks) {
      const int kb = ks * 32 + (half << 3);
#pragma unroll
      for (int nt = 0; nt < 8; ++nt) {
        FragBF kf;
        const int key = nt * 16 + r;
        kf.h[0] = *(const ushort8v*)&Kp[key * DK + kb];
        kf.h[1] = *(const ushort8v*)&Kp[key * DK + kb + 16];
        sc[nt].v = wmma_bf16(qf[ks].v, kf.v, sc[nt].v);
      }
    }

    // ---- online softmax (row = e + 8*half; 16-lane reductions) ----
    float nm[8];
#pragma unroll
    for (int e = 0; e < 8; ++e) nm[e] = rmax[e];
#pragma unroll
    for (int nt = 0; nt < 8; ++nt)
#pragma unroll
      for (int e = 0; e < 8; ++e) {
        sc[nt].f[e] *= scale;
        nm[e] = fmaxf(nm[e], sc[nt].f[e]);
      }
#pragma unroll
    for (int e = 0; e < 8; ++e) {
      nm[e] = fmaxf(nm[e], __shfl_xor(nm[e], 1, 32));
      nm[e] = fmaxf(nm[e], __shfl_xor(nm[e], 2, 32));
      nm[e] = fmaxf(nm[e], __shfl_xor(nm[e], 4, 32));
      nm[e] = fmaxf(nm[e], __shfl_xor(nm[e], 8, 32));
    }
    float alpha[8], part[8];
#pragma unroll
    for (int e = 0; e < 8; ++e) {
      alpha[e] = __expf(rmax[e] - nm[e]);
      rmax[e]  = nm[e];
      part[e]  = 0.f;
    }
#pragma unroll
    for (int nt = 0; nt < 8; ++nt)
#pragma unroll
      for (int e = 0; e < 8; ++e) {
        const float p = __expf(sc[nt].f[e] - nm[e]);
        sc[nt].f[e] = p;
        part[e] += p;
      }
#pragma unroll
    for (int e = 0; e < 8; ++e) {
      part[e] += __shfl_xor(part[e], 1, 32);
      part[e] += __shfl_xor(part[e], 2, 32);
      part[e] += __shfl_xor(part[e], 4, 32);
      part[e] += __shfl_xor(part[e], 8, 32);
      rsum[e] = rsum[e] * alpha[e] + part[e];
    }
#pragma unroll
    for (int dt = 0; dt < 4; ++dt)
#pragma unroll
      for (int e = 0; e < 8; ++e) o[dt].f[e] *= alpha[e];

    // ---- transpose P: C-layout regs -> A-layout via wave-private LDS ----
    __syncthreads();
#pragma unroll
    for (int nt = 0; nt < 8; ++nt)
#pragma unroll
      for (int e = 0; e < 8; ++e)
        Pw[(e + half * 8) * 128 + nt * 16 + r] = f2bf(sc[nt].f[e]);
    __syncthreads();

    // ---- O += P @ V  (K-dim = 128 keys -> 4 K-steps) ----
    const unsigned short* Vp = Vt + (size_t)bh * DK * SEQ + kt;
#pragma unroll
    for (int ks = 0; ks < 4; ++ks) {
      const int kb = ks * 32 + (half << 3);
      FragBF pf;
      pf.h[0] = *(const ushort8v*)&Pw[r * 128 + kb];
      pf.h[1] = *(const ushort8v*)&Pw[r * 128 + kb + 16];
#pragma unroll
      for (int dt = 0; dt < 4; ++dt) {
        FragBF vf;
        const int d = dt * 16 + r;
        vf.h[0] = *(const ushort8v*)&Vp[(size_t)d * SEQ + kb];
        vf.h[1] = *(const ushort8v*)&Vp[(size_t)d * SEQ + kb + 16];
        o[dt].v = wmma_bf16(pf.v, vf.v, o[dt].v);
      }
    }
  }

  // ---- normalize and write ctx as bf16 [B, S, DMODEL] (heads concatenated) ----
#pragma unroll
  for (int dt = 0; dt < 4; ++dt)
#pragma unroll
    for (int e = 0; e < 8; ++e) {
      const float val = o[dt].f[e] / rsum[e];
      const int s   = qbase + wave * 16 + e + half * 8;
      const int col = h * DK + dt * 16 + r;
      ctx[((size_t)b * SEQ + s) * DMODEL + col] = f2bf(val);
    }
}

// ---------------------------------------------------------------------------
extern "C" void kernel_launch(void* const* d_in, const int* in_sizes, int n_in,
                              void* d_out, int out_size, void* d_ws, size_t ws_size,
                              hipStream_t stream) {
  const float* q  = (const float*)d_in[0];
  const float* k  = (const float*)d_in[1];
  const float* v  = (const float*)d_in[2];
  const float* Wq = (const float*)d_in[3];
  const float* bq = (const float*)d_in[4];
  const float* Wk = (const float*)d_in[5];
  const float* bk = (const float*)d_in[6];
  const float* Wv = (const float*)d_in[7];
  const float* bv = (const float*)d_in[8];
  const float* Wo = (const float*)d_in[9];
  const float* bo = (const float*)d_in[10];

  unsigned short* ws = (unsigned short*)d_ws;
  const size_t HSZ = (size_t)BATCH * HEADS * SEQ * DK;   // 8388608 elements
  unsigned short* Qh  = ws;              // bf16 [B,H,S,DK]
  unsigned short* Kh  = ws + HSZ;        // bf16 [B,H,S,DK]
  unsigned short* Vt  = ws + 2 * HSZ;    // bf16 [B,H,DK,S]
  unsigned short* ctx = ws + 3 * HSZ;    // bf16 [B,S,DMODEL]

  dim3 blk(256);
  dim3 gproj(BATCH * SEQ / 128, DMODEL / 128);   // (64, 8)
  dim3 gattn(SEQ / 128, BATCH * HEADS);          // (16, 64)

  gemm_wmma_kernel<0, false><<<gproj, blk, 0, stream>>>(q, Wq, bq, Qh);
  gemm_wmma_kernel<0, false><<<gproj, blk, 0, stream>>>(k, Wk, bk, Kh);
  gemm_wmma_kernel<2, false><<<gproj, blk, 0, stream>>>(v, Wv, bv, Vt);
  flash_attn_kernel<<<gattn, blk, 0, stream>>>(Qh, Kh, Vt, ctx);
  gemm_wmma_kernel<1, true><<<gproj, blk, 0, stream>>>(ctx, Wo, bo, d_out);
}
